// GAT_17025250361353
// MI455X (gfx1250) — compile-verified, matrix operand order checked by reading, not run on hardware
//
#include <hip/hip_runtime.h>
#include <hip/hip_bf16.h>
#include <cstdint>
#include <cstddef>

#define NNODES 50000
#define NEDGES 800000
#define FIN    256
#define FH     128
#define FOUT   2

typedef __attribute__((ext_vector_type(16))) __bf16 v16bf;
typedef __attribute__((ext_vector_type(8)))  float  v8f;
typedef __attribute__((ext_vector_type(4)))  float  fl4;

// ---------- ordered-uint encoding for float atomic max ----------
__device__ __forceinline__ unsigned ord_f32(float f) {
  unsigned u = __float_as_uint(f);
  return (u & 0x80000000u) ? ~u : (u | 0x80000000u);
}
__device__ __forceinline__ float deord_f32(unsigned o) {
  unsigned b = (o & 0x80000000u) ? (o & 0x7fffffffu) : ~o;
  return __uint_as_float(b);
}
#define NEG_INF_ORD 0x007FFFFFu   // ord(-inf)

// ---------- pack W[K,Hout] (row-major f32) into WMMA B-fragment bf16 layout ----------
// layout: for tile t = kt*NT + nt, lane L, element i (0..15):
//   k = kt*32 + (L>>4)*8 + (i&7) + (i>>3)*16 ,  n = nt*16 + (L&15)
__global__ void pack_w_bf16(const float* __restrict__ W, __bf16* __restrict__ out,
                            int K, int Hout, int NT) {
  int idx = blockIdx.x * blockDim.x + threadIdx.x;
  int total = (K >> 5) * NT * 512;
  if (idx >= total) return;
  int i    = idx & 15;
  int lane = (idx >> 4) & 31;
  int t    = idx >> 9;
  int nt   = t % NT;
  int kt   = t / NT;
  int k = (kt << 5) + ((lane >> 4) << 3) + (i & 7) + ((i >> 3) << 4);
  int n = (nt << 4) + (lane & 15);
  float v = (n < Hout) ? W[(size_t)k * Hout + n] : 0.0f;
  out[idx] = (__bf16)v;
}

// ---------- A fragment: lane holds A[m][k], m=lane&15, k in {kb..kb+7, kb+16..kb+23}, kb=(lane>>4)*8 ----------
__device__ __forceinline__ v16bf load_a_frag(const float* __restrict__ arow, int kb, int reluA) {
  float va[16];
  *(fl4*)(va + 0)  = *(const fl4*)(arow + kb);
  *(fl4*)(va + 4)  = *(const fl4*)(arow + kb + 4);
  *(fl4*)(va + 8)  = *(const fl4*)(arow + kb + 16);
  *(fl4*)(va + 12) = *(const fl4*)(arow + kb + 20);
  v16bf a;
#pragma unroll
  for (int i = 0; i < 16; ++i) {
    float v = va[i];
    if (reluA) v = v > 0.0f ? v : 0.0f;
    a[i] = (__bf16)v;
  }
  return a;
}

// ---------- WMMA GEMM: D[M,Hout] = op(A[M,K]) @ W + bias1 + bias2 ----------
// one wave per (16-row strip, NB n-tiles). M%16==0, K%32==0.
template <int NB>
__global__ void gemm_wmma(const float* __restrict__ A, const __bf16* __restrict__ Bp,
                          const float* __restrict__ bias1, const float* __restrict__ bias2,
                          float* __restrict__ D, int M, int K, int Hout, int NT, int reluA) {
  int wave = (blockIdx.x * blockDim.x + threadIdx.x) >> 5;
  int lane = threadIdx.x & 31;
  int nBlk = NT / NB;
  int numTiles = (M >> 4) * nBlk;       // wave-uniform
  if (wave >= numTiles) return;         // uniform exit: EXEC stays all-ones for WMMA
  int nb = wave % nBlk;
  int mt = wave / nBlk;
  int m  = (mt << 4) + (lane & 15);
  int kbase = (lane >> 4) << 3;
  const float* arow = A + (size_t)m * K;
  int KT = K >> 5;

  v8f c[NB] = {};
  for (int kt = 0; kt < KT; ++kt) {
    int kb = (kt << 5) + kbase;
    __builtin_prefetch(arow + kb + 32, 0, 0);      // global_prefetch_b8
    v16bf a = load_a_frag(arow, kb, reluA);
#pragma unroll
    for (int q = 0; q < NB; ++q) {
      int nt = nb * NB + q;
      v16bf b = *(const v16bf*)(Bp + (((size_t)kt * NT + nt) * 32 + lane) * 16);
      c[q] = __builtin_amdgcn_wmma_f32_16x16x32_bf16(
          false, a, false, b, (short)0, c[q], false, false);
    }
  }

  // D layout: lanes 0-15: VGPR r -> (m0+r, n0+lane); lanes 16-31: (m0+8+r, n0+lane-16)
  int mrow = (mt << 4) + ((lane >> 4) << 3);
#pragma unroll
  for (int q = 0; q < NB; ++q) {
    int n = (nb * NB + q) * 16 + (lane & 15);
    if (n < Hout) {
      float badd = 0.0f;
      if (bias1) badd += bias1[n];
      if (bias2) badd += bias2[n];
#pragma unroll
      for (int r = 0; r < 8; ++r)
        D[(size_t)(mrow + r) * Hout + n] = c[q][r] + badd;
    }
  }
}

// ---------- segment-softmax init ----------
__global__ void seg_init(unsigned* __restrict__ mord, float* __restrict__ ssum, int n) {
  int i = blockIdx.x * blockDim.x + threadIdx.x;
  if (i < n) { mord[i] = NEG_INF_ORD; ssum[i] = 0.0f; }
}

// ---------- edge logit + segment max, H=128 (wave per edge) ----------
__global__ void edge_logit_max128(const int* __restrict__ src, const int* __restrict__ dst,
                                  const float* __restrict__ xs, const float* __restrict__ xd,
                                  const float* __restrict__ att,
                                  float* __restrict__ e, unsigned* __restrict__ mord, int E) {
  int edge = blockIdx.x * (blockDim.x >> 5) + (threadIdx.x >> 5);
  int lane = threadIdx.x & 31;
  if (edge >= E) return;
  int s = src[edge], d = dst[edge];
  fl4 a = *(const fl4*)(xs + (size_t)s * FH + lane * 4);
  fl4 b = *(const fl4*)(xd + (size_t)d * FH + lane * 4);
  fl4 w = *(const fl4*)(att + lane * 4);
  float sum = 0.0f;
#pragma unroll
  for (int j = 0; j < 4; ++j) {
    float v = a[j] + b[j];
    v = v > 0.0f ? v : 0.2f * v;          // LeakyReLU(0.2)
    sum += v * w[j];
  }
#pragma unroll
  for (int off = 16; off > 0; off >>= 1) sum += __shfl_xor(sum, off, 32);
  if (lane == 0) {
    e[edge] = sum;
    atomicMax(mord + d, ord_f32(sum));
  }
}

// ---------- edge logit + segment max, H=2 (thread per edge) ----------
__global__ void edge_logit_max2(const int* __restrict__ src, const int* __restrict__ dst,
                                const float* __restrict__ xs, const float* __restrict__ xd,
                                const float* __restrict__ att,
                                float* __restrict__ e, unsigned* __restrict__ mord, int E) {
  int i = blockIdx.x * blockDim.x + threadIdx.x;
  if (i >= E) return;
  int s = src[i], d = dst[i];
  float sum = 0.0f;
#pragma unroll
  for (int j = 0; j < FOUT; ++j) {
    float v = xs[(size_t)s * FOUT + j] + xd[(size_t)d * FOUT + j];
    v = v > 0.0f ? v : 0.2f * v;
    sum += v * att[j];
  }
  e[i] = sum;
  atomicMax(mord + d, ord_f32(sum));
}

// ---------- p = exp(e - m[dst]); s[dst] += p ----------
__global__ void edge_exp_sum(const int* __restrict__ dst, const unsigned* __restrict__ mord,
                             float* __restrict__ e, float* __restrict__ ssum, int E) {
  int i = blockIdx.x * blockDim.x + threadIdx.x;
  if (i >= E) return;
  int d = dst[i];
  float p = __expf(e[i] - deord_f32(mord[d]));
  e[i] = p;
  atomicAdd(ssum + d, p);
}

// ---------- out[dst] += alpha * xs[src], H=128 (wave per edge) ----------
__global__ void edge_scatter128(const int* __restrict__ src, const int* __restrict__ dst,
                                const float* __restrict__ xs, const float* __restrict__ p,
                                const float* __restrict__ ssum, float* __restrict__ out, int E) {
  int edge = blockIdx.x * (blockDim.x >> 5) + (threadIdx.x >> 5);
  int lane = threadIdx.x & 31;
  if (edge >= E) return;
  int s = src[edge], d = dst[edge];
  float alpha = p[edge] / (ssum[d] + 1e-16f);
  fl4 a = *(const fl4*)(xs + (size_t)s * FH + lane * 4);
  float* op = out + (size_t)d * FH + lane * 4;
#pragma unroll
  for (int j = 0; j < 4; ++j) atomicAdd(op + j, alpha * a[j]);
}

// ---------- H=2 scatter (thread per edge) ----------
__global__ void edge_scatter2(const int* __restrict__ src, const int* __restrict__ dst,
                              const float* __restrict__ xs, const float* __restrict__ p,
                              const float* __restrict__ ssum, float* __restrict__ out, int E) {
  int i = blockIdx.x * blockDim.x + threadIdx.x;
  if (i >= E) return;
  int s = src[i], d = dst[i];
  float alpha = p[i] / (ssum[d] + 1e-16f);
#pragma unroll
  for (int j = 0; j < FOUT; ++j)
    atomicAdd(out + (size_t)d * FOUT + j, alpha * xs[(size_t)s * FOUT + j]);
}

// ---------- elementwise relu ----------
__global__ void relu_k(const float* __restrict__ in, float* __restrict__ out, int n) {
  int i = blockIdx.x * blockDim.x + threadIdx.x;
  if (i < n) { float v = in[i]; out[i] = v > 0.0f ? v : 0.0f; }
}

// ================= host-side helpers =================
static void run_gemm(const float* A, const float* W, const float* b1, const float* b2,
                     float* D, int M, int K, int Hout, int reluA,
                     __bf16* packB, hipStream_t stream) {
  int NT = (Hout + 15) / 16;
  int total = (K >> 5) * NT * 512;
  pack_w_bf16<<<(total + 255) / 256, 256, 0, stream>>>(W, packB, K, Hout, NT);
  if (NT % 4 == 0) {
    int tiles = (M >> 4) * (NT / 4);
    gemm_wmma<4><<<(tiles + 7) / 8, 256, 0, stream>>>(A, packB, b1, b2, D, M, K, Hout, NT, reluA);
  } else {
    int tiles = (M >> 4) * NT;
    gemm_wmma<1><<<(tiles + 7) / 8, 256, 0, stream>>>(A, packB, b1, b2, D, M, K, Hout, NT, reluA);
  }
}

static void run_edge_phase(const int* esrc, const int* edst,
                           const float* xs, const float* xd, const float* att,
                           float* acc, float* e, float* ssum, unsigned* mord,
                           int Hdim, hipStream_t stream) {
  seg_init<<<(NNODES + 255) / 256, 256, 0, stream>>>(mord, ssum, NNODES);
  if (Hdim == FH) {
    edge_logit_max128<<<NEDGES / 8, 256, 0, stream>>>(esrc, edst, xs, xd, att, e, mord, NEDGES);
  } else {
    edge_logit_max2<<<(NEDGES + 255) / 256, 256, 0, stream>>>(esrc, edst, xs, xd, att, e, mord, NEDGES);
  }
  edge_exp_sum<<<(NEDGES + 255) / 256, 256, 0, stream>>>(edst, mord, e, ssum, NEDGES);
  if (Hdim == FH) {
    edge_scatter128<<<NEDGES / 8, 256, 0, stream>>>(esrc, edst, xs, e, ssum, acc, NEDGES);
  } else {
    edge_scatter2<<<(NEDGES + 255) / 256, 256, 0, stream>>>(esrc, edst, xs, e, ssum, acc, NEDGES);
  }
}

extern "C" void kernel_launch(void* const* d_in, const int* in_sizes, int n_in,
                              void* d_out, int out_size, void* d_ws, size_t ws_size,
                              hipStream_t stream) {
  (void)in_sizes; (void)n_in; (void)out_size; (void)ws_size;
  const float* x      = (const float*)d_in[0];
  const int*   ei     = (const int*)d_in[1];
  const int*   esrc   = ei;
  const int*   edst   = ei + NEDGES;
  const float* c0_Wl  = (const float*)d_in[2];
  const float* c0_Wr  = (const float*)d_in[3];
  const float* c0_att = (const float*)d_in[4];
  const float* c0_b   = (const float*)d_in[5];
  const float* c1_Wl  = (const float*)d_in[6];
  const float* c1_Wr  = (const float*)d_in[7];
  const float* c1_att = (const float*)d_in[8];
  const float* c1_b   = (const float*)d_in[9];
  const float* cf_Wl  = (const float*)d_in[10];
  const float* cf_Wr  = (const float*)d_in[11];
  const float* cf_att = (const float*)d_in[12];
  const float* cf_b   = (const float*)d_in[13];
  const float* l0_W   = (const float*)d_in[14];
  const float* l0_b   = (const float*)d_in[15];
  const float* l1_W   = (const float*)d_in[16];
  const float* l1_b   = (const float*)d_in[17];
  const float* lf_W   = (const float*)d_in[18];
  const float* lf_b   = (const float*)d_in[19];

  float* out_logits = (float*)d_out;                       // [N,2]
  float* emb        = (float*)d_out + (size_t)NNODES * FOUT; // [N,128]

  // workspace carve (all chunks 64B aligned by construction)
  char* w = (char*)d_ws;
  float*    xs    = (float*)w;              w += (size_t)NNODES * FH * 4;
  float*    xd    = (float*)w;              w += (size_t)NNODES * FH * 4;
  float*    acc   = (float*)w;              w += (size_t)NNODES * FH * 4;
  float*    h     = (float*)w;              w += (size_t)NNODES * FH * 4;
  float*    e     = (float*)w;              w += (size_t)NEDGES * 4;
  float*    ssum  = (float*)w;              w += (size_t)NNODES * 4;
  unsigned* mord  = (unsigned*)w;           w += (size_t)NNODES * 4;
  __bf16*   packB = (__bf16*)w;             // <= 256*128 bf16 = 64KB

  // ---- Layer 0: x(IN=256) -> h(H=128) ----
  run_gemm(x, c0_Wl, nullptr, nullptr, xs,  NNODES, FIN, FH, 0, packB, stream);
  run_gemm(x, c0_Wr, nullptr, nullptr, xd,  NNODES, FIN, FH, 0, packB, stream);
  run_gemm(x, l0_W,  l0_b,    c0_b,    acc, NNODES, FIN, FH, 1, packB, stream); // res + both biases
  run_edge_phase(esrc, edst, xs, xd, c0_att, acc, e, ssum, mord, FH, stream);
  relu_k<<<(NNODES * FH + 255) / 256, 256, 0, stream>>>(acc, h, NNODES * FH);

  // ---- Layer 1: h -> emb (written directly into d_out emb region) ----
  run_gemm(h, c1_Wl, nullptr, nullptr, xs,  NNODES, FH, FH, 0, packB, stream);
  run_gemm(h, c1_Wr, nullptr, nullptr, xd,  NNODES, FH, FH, 0, packB, stream);
  run_gemm(h, l1_W,  l1_b,    c1_b,    emb, NNODES, FH, FH, 1, packB, stream);
  run_edge_phase(esrc, edst, xs, xd, c1_att, emb, e, ssum, mord, FH, stream);

  // ---- Final layer: emb -> logits (OUT=2, padded to one 16-wide n-tile) ----
  run_gemm(emb, cf_Wl, nullptr, nullptr, xs,         NNODES, FH, FOUT, 0, packB, stream);
  run_gemm(emb, cf_Wr, nullptr, nullptr, xd,         NNODES, FH, FOUT, 0, packB, stream);
  run_gemm(emb, lf_W,  lf_b,    cf_b,    out_logits, NNODES, FH, FOUT, 1, packB, stream);
  run_edge_phase(esrc, edst, xs, xd, cf_att, out_logits, e, ssum, mord, FOUT, stream);
}
